// DeformableStateSpaceModel_87995289961155
// MI455X (gfx1250) — matrix operand order repaired and checked by reading, not compiled, not run on hardware
//
#include <hip/hip_runtime.h>
#include <hip/hip_bf16.h>
#include <cstdint>

// ---------------------------------------------------------------------------
// DeformableStateSpaceModel forward for gfx1250 (MI455X), wave32.
// GEMMs (SSM in-proj / out-proj) run on v_wmma_f32_16x16x32_f16 with
// fragment-packed B (contiguous 32B loads/lane) and A reuse across an
// 8-tile N strip (32 WMMAs per wave).
// ---------------------------------------------------------------------------

#define BN   8
#define CCH  128
#define HH   64
#define WW   32
#define HW   2048              // H*W == seq length L
#define MTOK (BN*HW)           // 16384 tokens
#define KDIM 128               // GEMM K (= C)
#define KS   (KDIM/32)         // 4 k-steps
#define NT   8                 // n-tiles per wave (A reuse factor)
#define DI   256
#define RR   8

typedef _Float16 v16h __attribute__((ext_vector_type(16)));
typedef _Float16 v8h  __attribute__((ext_vector_type(8)));
typedef float    v8f  __attribute__((ext_vector_type(8)));

__device__ __forceinline__ float sigf(float x)  { return 1.f/(1.f+expf(-x)); }
__device__ __forceinline__ float siluf(float x) { return x*sigf(x); }
__device__ __forceinline__ float softplusf(float x){
  return (x > 0.f) ? x + log1pf(expf(-x)) : log1pf(expf(x));
}
__device__ __forceinline__ float wred(float v){
  #pragma unroll
  for (int o = 16; o; o >>= 1) v += __shfl_xor(v, o, 32);
  return v;
}

// ---------------- depthwise 3x3 conv (SAME, cross-correlation) -------------
__global__ void k_dwconv(const float* __restrict__ x, const float* __restrict__ w,
                         const float* __restrict__ bias, float* __restrict__ out,
                         int do_silu)
{
  int idx = blockIdx.x*blockDim.x + threadIdx.x;
  if (idx >= BN*CCH*HH*WW) return;
  int wi = idx & 31, hi = (idx >> 5) & 63, c = (idx >> 11) & 127;
  const float* xp = x + (idx >> 11) * (HH*WW);   // (b,c) plane
  const float* wp = w + c*9;
  float acc = bias[c];
  #pragma unroll
  for (int kh = 0; kh < 3; ++kh) {
    int yy = hi + kh - 1;
    if (yy < 0 || yy >= HH) continue;
    #pragma unroll
    for (int kw = 0; kw < 3; ++kw) {
      int xx = wi + kw - 1;
      if (xx < 0 || xx >= WW) continue;
      acc += xp[yy*WW + xx] * wp[kh*3 + kw];
    }
  }
  out[idx] = do_silu ? siluf(acc) : acc;
}

// ---------------- channel mean over (h,w) ----------------------------------
__global__ void k_chan_mean(const float* __restrict__ x, float* __restrict__ mean)
{
  __shared__ float red[256];
  int bc = blockIdx.x;
  float s = 0.f;
  for (int i = threadIdx.x; i < HW; i += 256) s += x[bc*HW + i];
  red[threadIdx.x] = s; __syncthreads();
  for (int o = 128; o; o >>= 1) {
    if ((int)threadIdx.x < o) red[threadIdx.x] += red[threadIdx.x + o];
    __syncthreads();
  }
  if (threadIdx.x == 0) mean[bc] = red[0] * (1.f/(float)HW);
}

// ---------------- SE MLP: 128 -> 8 (relu) -> 128 (sigmoid) -----------------
__global__ void k_se(const float* __restrict__ mean,
                     const float* __restrict__ f1w, const float* __restrict__ f1b,
                     const float* __restrict__ f2w, const float* __restrict__ f2b,
                     float* __restrict__ se)
{
  __shared__ float y1[8];
  int b = blockIdx.x, t = threadIdx.x;
  if (t < 8) {
    float a = f1b[t];
    for (int c = 0; c < CCH; ++c) a += mean[b*CCH + c] * f1w[t*CCH + c];
    y1[t] = fmaxf(a, 0.f);
  }
  __syncthreads();
  float a = f2b[t];
  #pragma unroll
  for (int j = 0; j < 8; ++j) a += y1[j] * f2w[t*8 + j];
  se[b*CCH + t] = sigf(a);
}

// ------- SE-scale, exact GELU, channel-LN, 1x1 conv (3ch), tanh, offsets ---
__global__ void k_offset_post(const float* __restrict__ xoff, const float* __restrict__ se,
                              const float* __restrict__ lng, const float* __restrict__ lnb,
                              const float* __restrict__ c1w,
                              float* __restrict__ dpx, float* __restrict__ dpy,
                              float* __restrict__ didx)
{
  int wave = (blockIdx.x*blockDim.x + threadIdx.x) >> 5;
  int lane = threadIdx.x & 31;
  if (wave >= BN*HW) return;
  int b = wave >> 11, pix = wave & 2047;
  float v[4];
  #pragma unroll
  for (int q = 0; q < 4; ++q) {
    int c = lane + 32*q;
    float x = xoff[(b*CCH + c)*HW + pix] * se[b*CCH + c];
    v[q] = 0.5f * x * (1.f + erff(x * 0.70710678118654752f)); // exact GELU
  }
  float s  = v[0]+v[1]+v[2]+v[3];
  float s2 = v[0]*v[0]+v[1]*v[1]+v[2]*v[2]+v[3]*v[3];
  s = wred(s); s2 = wred(s2);
  float mu  = s * (1.f/128.f);
  float var = s2 * (1.f/128.f) - mu*mu;
  float inv = rsqrtf(var + 1e-6f);
  float o0 = 0.f, o1 = 0.f, o2 = 0.f;
  #pragma unroll
  for (int q = 0; q < 4; ++q) {
    int c = lane + 32*q;
    float xn = (v[q]-mu)*inv*lng[c] + lnb[c];
    o0 += c1w[0*CCH + c]*xn;
    o1 += c1w[1*CCH + c]*xn;
    o2 += c1w[2*CCH + c]*xn;
  }
  o0 = wred(o0); o1 = wred(o1); o2 = wred(o2);
  if (lane == 0) {
    dpx[wave]  = tanhf(o0) * (1.f/(float)WW);
    dpy[wave]  = tanhf(o1) * (1.f/(float)HH);
    didx[wave] = (float)pix * (2.f/(float)(HW-1)) - 1.f + tanhf(o2);
  }
}

// ---------------- bilinear grid sample (zero outside) ----------------------
__global__ void k_grid_sample(const float* __restrict__ xdw,
                              const float* __restrict__ dpx, const float* __restrict__ dpy,
                              float* __restrict__ out)
{
  int idx = blockIdx.x*blockDim.x + threadIdx.x;
  if (idx >= BN*CCH*HW) return;
  int pix = idx & 2047;
  int b   = idx >> 18;
  int hi = pix >> 5, wi = pix & 31;
  float gx = (-1.f + wi*(2.f/31.f) + dpx[b*HW + pix] + 1.f) * (WW*0.5f) - 0.5f;
  float gy = (-1.f + hi*(2.f/63.f) + dpy[b*HW + pix] + 1.f) * (HH*0.5f) - 0.5f;
  float x0f = floorf(gx), y0f = floorf(gy);
  int   x0  = (int)x0f,  y0  = (int)y0f;
  float wx = gx - x0f, wy = gy - y0f;
  const float* xf = xdw + (idx >> 11) * HW;  // (b,c) plane
  float acc = 0.f;
  #pragma unroll
  for (int dy = 0; dy < 2; ++dy)
    #pragma unroll
    for (int dx = 0; dx < 2; ++dx) {
      int yy = y0 + dy, xx = x0 + dx;
      float wgt = (dx ? wx : 1.f-wx) * (dy ? wy : 1.f-wy);
      bool valid = (xx >= 0) & (xx < WW) & (yy >= 0) & (yy < HH);
      int yc = yy < 0 ? 0 : (yy > HH-1 ? HH-1 : yy);
      int xc = xx < 0 ? 0 : (xx > WW-1 ? WW-1 : xx);
      acc += xf[yc*WW + xc] * (valid ? wgt : 0.f);
    }
  out[idx] = acc;
}

// ---------------- stable argsort of 2048 keys (bitonic, per batch) ---------
__global__ void k_argsort(const float* __restrict__ keys, int* __restrict__ sidx)
{
  __shared__ float sk[HW];
  __shared__ int   sv[HW];
  int b = blockIdx.x;
  for (int i = threadIdx.x; i < HW; i += 1024) { sk[i] = keys[b*HW + i]; sv[i] = i; }
  __syncthreads();
  for (int k = 2; k <= HW; k <<= 1) {
    for (int j = k >> 1; j > 0; j >>= 1) {
      int t = threadIdx.x;                 // 1024 threads, one compare each
      int i = ((t & ~(j-1)) << 1) | (t & (j-1));
      int p = i | j;
      bool up = ((i & k) == 0);
      float ki = sk[i], kp = sk[p];
      int   vi = sv[i], vp = sv[p];
      bool gt = (ki > kp) || (ki == kp && vi > vp);  // lexicographic -> stable
      if (gt == up) { sk[i]=kp; sk[p]=ki; sv[i]=vp; sv[p]=vi; }
      __syncthreads();
    }
  }
  for (int i = threadIdx.x; i < HW; i += 1024) sidx[b*HW + i] = sv[i];
}

// ---- build f16 token-major sequence (mode 0 fwd, 1 h-reversed, 2 gather) --
__global__ void k_pack_seq(const float* __restrict__ src, const int* __restrict__ sidx,
                           _Float16* __restrict__ X, int mode)
{
  int idx = blockIdx.x*blockDim.x + threadIdx.x;
  if (idx >= MTOK*CCH) return;
  int c = idx & 127, l = (idx >> 7) & 2047, b = idx >> 18;
  int l2 = l;
  if (mode == 1) { int h = l >> 5, w = l & 31; l2 = (HH-1-h)*WW + w; }
  else if (mode == 2) { l2 = sidx[b*HW + l]; }
  X[idx] = (_Float16)src[(b*CCH + c)*HW + l2];
}

// ---- pack weight (O=N rows, I=K cols) into per-lane WMMA B fragments ------
// Bp[((nt*KS + ks)*32 + lane)*16 + e] = (f16) W[n*KDIM + k]
//   with k = ks*32 + 16*(lane>>4) + e,  n = nt*16 + (lane&15)
// so the GEMM inner loop reads one contiguous 32B fragment per lane.
__global__ void k_pack_B(const float* __restrict__ Wsrc, _Float16* __restrict__ Bp, int N)
{
  int idx = blockIdx.x*blockDim.x + threadIdx.x;
  if (idx >= N*KDIM) return;
  int e    = idx & 15;
  int lane = (idx >> 4) & 31;
  int ks   = (idx >> 9) & (KS-1);
  int nt   = idx >> 11;
  int k = ks*32 + ((lane >> 4) << 4) + e;
  int n = (nt << 4) + (lane & 15);
  Bp[idx] = (_Float16)Wsrc[n*KDIM + k];
}

// ---------------- WMMA GEMM: C[M,N] = A[M,128] @ B[128,N] + bias -----------
// One wave owns a 16 x (16*NT) strip: A fragment loaded once per k-step and
// reused across NT WMMAs (32 v_wmma_f32_16x16x32_f16 per wave).
// Fragment layouts per CDNA5 ISA 7.12.2 (wave32):
//   A: row = lane%16; elems 0..15 -> K = k0 + 16*(lane/16) + e (contiguous,
//                     same k order as k_pack_B, not the ISA table order)
//   B: pre-packed (k_pack_B) -> contiguous v16h per lane
//   C/D: col = lane%16; vgpr v -> row = 8*(lane/16) + v
__global__ void k_gemm_wmma(const _Float16* __restrict__ A, const _Float16* __restrict__ Bp,
                            const float* __restrict__ bias, float* __restrict__ Cmat,
                            int M, int N, int accMode)
{
  int wave = (blockIdx.x*blockDim.x + threadIdx.x) >> 5;
  int lane = threadIdx.x & 31;
  int stripsN = (N >> 4) / NT;
  int numStrips = (M >> 4) * stripsN;
  if (wave >= numStrips) return;
  int m0     = (wave / stripsN) << 4;
  int ntBase = (wave % stripsN) * NT;
  int row    = m0 + (lane & 15);
  int khalfA = (lane >> 4) << 4;   // 0 or 16

  v8f zero = {};
  v8f acc[NT];
  #pragma unroll
  for (int t = 0; t < NT; ++t) acc[t] = zero;

  #pragma unroll
  for (int ks = 0; ks < KS; ++ks) {
    const _Float16* ap = A + (size_t)row*KDIM + ks*32 + khalfA;
    v8h alo = *(const v8h*)(ap);
    v8h ahi = *(const v8h*)(ap + 8);
    v16h av = __builtin_shufflevector(alo, ahi,
                0,1,2,3,4,5,6,7,8,9,10,11,12,13,14,15);
    #pragma unroll
    for (int t = 0; t < NT; ++t) {
      const v16h* bptr = (const v16h*)
          (Bp + ((((size_t)(ntBase + t)*KS + ks)*32 + lane) << 4));
      v16h bv = *bptr;
      acc[t] = __builtin_amdgcn_wmma_f32_16x16x32_f16(
          /*neg_a=*/false, av, /*neg_b=*/false, bv,
          /*c_mod=*/(short)0, acc[t], /*reuse_a=*/false, /*reuse_b=*/false);
    }
  }

  int rbase = m0 + ((lane >> 4) << 3);
  #pragma unroll
  for (int t = 0; t < NT; ++t) {
    int col = ((ntBase + t) << 4) + (lane & 15);
    float bb = bias[col];
    #pragma unroll
    for (int v = 0; v < 8; ++v) {
      int oidx = (rbase + v)*N + col;
      float val = acc[t][v] + bb;
      if (accMode) Cmat[oidx] += val; else Cmat[oidx] = val;
    }
  }
}

// ---------------- dt projection: softplus(silu(xr[:,:256]) @ dt_w.T + b) ---
__global__ void k_dt(const float* __restrict__ xr, const float* __restrict__ dtw,
                     const float* __restrict__ dtb, float* __restrict__ dt)
{
  int idx = blockIdx.x*blockDim.x + threadIdx.x;
  if (idx >= MTOK*RR) return;
  int i = idx >> 3, j = idx & 7;
  const float* xri = xr + (size_t)i*512;
  float a = dtb[j];
  for (int k = 0; k < DI; ++k) a += siluf(xri[k]) * dtw[j*DI + k];
  dt[idx] = softplusf(a);
}

// ---------------- selective-scan: h_t = e^{dt*A}(h_{t-1} + x1*dt) ----------
__global__ void k_scan(const float* __restrict__ xr, const float* __restrict__ dt,
                       const float* __restrict__ Alog, const float* __restrict__ Dp,
                       float* __restrict__ y)
{
  int b = blockIdx.x, n = threadIdx.x;
  float Av[RR], h[RR];
  #pragma unroll
  for (int r = 0; r < RR; ++r) { Av[r] = -expf(Alog[r*CCH + n]); h[r] = 0.f; }
  float Dv = Dp[n];
  for (int l = 0; l < HW; ++l) {
    int i = b*HW + l;
    float x1 = siluf(xr[(size_t)i*512 + n]);
    float accv = 0.f;
    #pragma unroll
    for (int r = 0; r < RR; ++r) {
      float dtv = dt[i*RR + r];
      float e = expf(dtv * Av[r]);
      h[r] = e * (h[r] + x1*dtv);
      accv += h[r];
    }
    y[(size_t)i*CCH + n] = accv * (1.f/(float)RR) + x1*Dv;
  }
}

// ---------------- y * silu(res) -> f16 for out-proj ------------------------
__global__ void k_ygate(const float* __restrict__ y, const float* __restrict__ xr,
                        _Float16* __restrict__ Y)
{
  int idx = blockIdx.x*blockDim.x + threadIdx.x;
  if (idx >= MTOK*CCH) return;
  int i = idx >> 7, c = idx & 127;
  float rv = xr[(size_t)i*512 + 256 + c];
  Y[idx] = (_Float16)(y[idx] * siluf(rv));
}

// ---------------- width-mixing gate: silu(ms @ gate_w.T + b) ---------------
__global__ void k_gate(const float* __restrict__ ms, const float* __restrict__ gw,
                       const float* __restrict__ gb, float* __restrict__ gate)
{
  int idx = blockIdx.x*blockDim.x + threadIdx.x;
  if (idx >= BN*CCH*HH*WW) return;
  int o = idx & 31;
  const float* msr = ms + (idx & ~31);
  float acc = gb[o];
  #pragma unroll
  for (int w2 = 0; w2 < WW; ++w2) acc += gw[o*WW + w2] * msr[w2];
  gate[idx] = siluf(acc);
}

// ---------------- merge/3 * gate, channel-LN, NCHW out ---------------------
__global__ void k_final(const float* __restrict__ ssum, const float* __restrict__ gate,
                        const float* __restrict__ lng, const float* __restrict__ lnb,
                        float* __restrict__ out)
{
  int wave = (blockIdx.x*blockDim.x + threadIdx.x) >> 5;
  int lane = threadIdx.x & 31;
  if (wave >= BN*HW) return;
  int b = wave >> 11, pix = wave & 2047;
  int hi = pix >> 5, wi = pix & 31;
  float m[4];
  #pragma unroll
  for (int q = 0; q < 4; ++q) {
    int c = lane + 32*q;
    m[q] = ssum[(size_t)(b*HW + pix)*CCH + c] * (1.f/3.f)
         * gate[((b*CCH + c)*HH + hi)*WW + wi];
  }
  float s  = m[0]+m[1]+m[2]+m[3];
  float s2 = m[0]*m[0]+m[1]*m[1]+m[2]*m[2]+m[3]*m[3];
  s = wred(s); s2 = wred(s2);
  float mu  = s * (1.f/128.f);
  float var = s2 * (1.f/128.f) - mu*mu;
  float inv = rsqrtf(var + 1e-5f);
  #pragma unroll
  for (int q = 0; q < 4; ++q) {
    int c = lane + 32*q;
    out[((b*CCH + c)*HH + hi)*WW + wi] = (m[q]-mu)*inv*lng[c] + lnb[c];
  }
}

// ===========================================================================
extern "C" void kernel_launch(void* const* d_in, const int* in_sizes, int n_in,
                              void* d_out, int out_size, void* d_ws, size_t ws_size,
                              hipStream_t stream)
{
  (void)in_sizes; (void)n_in; (void)out_size; (void)ws_size;
  // Flattening assumes jax tree_leaves order (dict keys sorted at each level):
  // 0 ms, 1 pan, 2 dw_b, 3 dw_w, 4 gate_b, 5 gate_w, 6 ln_b, 7 ln_g,
  // 8..16 offset{c1_w,dw_b,dw_w,fc1_b,fc1_w,fc2_b,fc2_w,ln_b,ln_g},
  // 17..24 ssm_b{A_log,D,dt_b,dt_w,in_b,in_w,out_b,out_w}, 25..32 ssm_d, 33..40 ssm_f
  const float* ms     = (const float*)d_in[0];
  const float* dw_b   = (const float*)d_in[2];
  const float* dw_w   = (const float*)d_in[3];
  const float* gate_b = (const float*)d_in[4];
  const float* gate_w = (const float*)d_in[5];
  const float* ln_b   = (const float*)d_in[6];
  const float* ln_g   = (const float*)d_in[7];
  const float* o_c1w  = (const float*)d_in[8];
  const float* o_dwb  = (const float*)d_in[9];
  const float* o_dww  = (const float*)d_in[10];
  const float* o_f1b  = (const float*)d_in[11];
  const float* o_f1w  = (const float*)d_in[12];
  const float* o_f2b  = (const float*)d_in[13];
  const float* o_f2w  = (const float*)d_in[14];
  const float* o_lnb  = (const float*)d_in[15];
  const float* o_lng  = (const float*)d_in[16];
  const int SSM_B = 17, SSM_D = 25, SSM_F = 33;   // A_log,D,dt_b,dt_w,in_b,in_w,out_b,out_w

  // ---- workspace carving (256B aligned) ----
  uint8_t* base = (uint8_t*)d_ws;
  size_t off = 0;
  auto carve = [&](size_t bytes) -> void* {
    void* p = base + off;
    off = (off + bytes + 255) & ~(size_t)255;
    return p;
  };
  const size_t NELEM = (size_t)BN*CCH*HH*WW;           // 2,097,152
  float*    xdw   = (float*)carve(NELEM*4);
  float*    xoff  = (float*)carve(NELEM*4);
  float*    feat  = (float*)carve(NELEM*4);            // def_feat
  float*    gatep = (float*)carve(NELEM*4);
  float*    xr    = (float*)carve((size_t)MTOK*512*4);
  float*    ybuf  = (float*)carve((size_t)MTOK*CCH*4);
  float*    ssum  = (float*)carve((size_t)MTOK*CCH*4);
  _Float16* Xh    = (_Float16*)carve((size_t)MTOK*CCH*2);
  _Float16* Yh    = (_Float16*)carve((size_t)MTOK*CCH*2);
  _Float16* inwP  = (_Float16*)carve((size_t)512*KDIM*2);   // packed B, in-proj
  _Float16* outwP = (_Float16*)carve((size_t)CCH*KDIM*2);   // packed B, out-proj
  float*    cmean = (float*)carve(BN*CCH*4);
  float*    sebuf = (float*)carve(BN*CCH*4);
  float*    dpx   = (float*)carve((size_t)BN*HW*4);
  float*    dpy   = (float*)carve((size_t)BN*HW*4);
  float*    didx  = (float*)carve((size_t)BN*HW*4);
  int*      sidx  = (int*)carve((size_t)BN*HW*4);
  float*    dtbuf = (float*)carve((size_t)MTOK*RR*4);

  const int TPB = 256;
  const int nElemBlocks = (int)(NELEM / TPB);          // 8192

  // 1) x_dw = silu(dwconv(ms));  pan branch in reference is discarded.
  k_dwconv<<<nElemBlocks, TPB, 0, stream>>>(ms, dw_w, dw_b, xdw, 1);
  // 2) offset net
  k_dwconv<<<nElemBlocks, TPB, 0, stream>>>(xdw, o_dww, o_dwb, xoff, 0);
  k_chan_mean<<<BN*CCH, TPB, 0, stream>>>(xoff, cmean);
  k_se<<<BN, CCH, 0, stream>>>(cmean, o_f1w, o_f1b, o_f2w, o_f2b, sebuf);
  k_offset_post<<<(BN*HW*32)/TPB, TPB, 0, stream>>>(xoff, sebuf, o_lng, o_lnb,
                                                    o_c1w, dpx, dpy, didx);
  // 3) deformable sampling + ordering
  k_grid_sample<<<nElemBlocks, TPB, 0, stream>>>(xdw, dpx, dpy, feat);
  k_argsort<<<BN, 1024, 0, stream>>>(didx, sidx);

  // 4) three SSM blocks (deform writes ssum, fwd/bwd accumulate)
  struct Blk { int pbase; const float* src; int mode; int acc; };
  Blk blks[3] = { {SSM_D, feat, 2, 0}, {SSM_F, xdw, 0, 1}, {SSM_B, xdw, 1, 1} };
  for (int s = 0; s < 3; ++s) {
    const float* A_log = (const float*)d_in[blks[s].pbase + 0];
    const float* Dp    = (const float*)d_in[blks[s].pbase + 1];
    const float* dt_b  = (const float*)d_in[blks[s].pbase + 2];
    const float* dt_w  = (const float*)d_in[blks[s].pbase + 3];
    const float* in_b  = (const float*)d_in[blks[s].pbase + 4];
    const float* in_w  = (const float*)d_in[blks[s].pbase + 5];
    const float* out_b = (const float*)d_in[blks[s].pbase + 6];
    const float* out_w = (const float*)d_in[blks[s].pbase + 7];

    k_pack_seq<<<nElemBlocks, TPB, 0, stream>>>(blks[s].src, sidx, Xh, blks[s].mode);
    k_pack_B<<<(512*KDIM)/TPB, TPB, 0, stream>>>(in_w, inwP, 512);
    k_pack_B<<<(CCH*KDIM)/TPB, TPB, 0, stream>>>(out_w, outwP, CCH);
    // in-proj: (16384 x 128) @ (128 x 512)
    {
      int strips = (MTOK/16) * ((512/16)/NT);          // 4096 waves
      k_gemm_wmma<<<(strips*32 + TPB-1)/TPB, TPB, 0, stream>>>(Xh, inwP, in_b, xr,
                                                               MTOK, 512, 0);
    }
    k_dt<<<(MTOK*RR)/TPB, TPB, 0, stream>>>(xr, dt_w, dt_b, dtbuf);
    k_scan<<<BN, CCH, 0, stream>>>(xr, dtbuf, A_log, Dp, ybuf);
    k_ygate<<<nElemBlocks, TPB, 0, stream>>>(ybuf, xr, Yh);
    // out-proj: (16384 x 128) @ (128 x 128), accumulate into ssum
    {
      int strips = (MTOK/16) * ((CCH/16)/NT);          // 1024 waves
      k_gemm_wmma<<<(strips*32 + TPB-1)/TPB, TPB, 0, stream>>>(Yh, outwP, out_b, ssum,
                                                               MTOK, CCH, blks[s].acc);
    }
  }

  // 5) gate + merge + final LN
  k_gate<<<nElemBlocks, TPB, 0, stream>>>(ms, gate_w, gate_b, gatep);
  k_final<<<(BN*HW*32)/TPB, TPB, 0, stream>>>(ssum, gatep, ln_g, ln_b, (float*)d_out);
}
